// Attention_1202590843615
// MI455X (gfx1250) — compile-verified, matrix operand order checked by reading, not run on hardware
//
#include <hip/hip_runtime.h>
#include <hip/hip_bf16.h>

typedef __bf16 bf16;
typedef __attribute__((ext_vector_type(16))) __bf16   v16bf;
typedef __attribute__((ext_vector_type(8)))  float    v8f;
typedef __attribute__((ext_vector_type(4)))  unsigned int v4u;
typedef __attribute__((ext_vector_type(4)))  int v4i;

#define DIM     1024
#define NHEADS  16
#define HDIM    64
#define BATCH   2
#define SEQ     4096
#define M_TOTAL (BATCH * SEQ)   // 8192

// ---- async copy to LDS (gfx1250 GLOBAL_LOAD_ASYNC_TO_LDS_B128) ------------
#if defined(__has_builtin)
#if __has_builtin(__builtin_amdgcn_global_load_async_to_lds_b128) && \
    __has_builtin(__builtin_amdgcn_s_wait_asynccnt)
#define HAVE_ASYNC_LDS 1
#endif
#endif

#if defined(HAVE_ASYNC_LDS)
typedef __attribute__((address_space(1))) v4i g_v4i;   // global int4
typedef __attribute__((address_space(3))) v4i l_v4i;   // LDS int4
// Generic LDS pointer: low 32 bits are the LDS segment offset (aperture rule).
#define AS1(p) ((g_v4i*)(uintptr_t)(p))
#define AS3(p) ((l_v4i*)(unsigned int)(uintptr_t)(p))
#endif

// ---------------------------------------------------------------------------
// WMMA helper: D = A(16x32 bf16) x B(32x16 bf16) + C(16x16 f32)
// ---------------------------------------------------------------------------
__device__ __forceinline__ v8f wmma_bf16(v16bf a, v16bf b, v8f c) {
  return __builtin_amdgcn_wmma_f32_16x16x32_bf16(
      /*neg_a=*/false, a, /*neg_b=*/false, b,
      /*c_mod=*/(short)0, c, /*reuse_a=*/false, /*reuse_b=*/false);
}

// Load one 16x32 bf16 fragment (A-operand layout; B-operand uses the same
// lane pattern with "row" = output column index) from a K-contiguous,
// row-major bf16 buffer (global or LDS).
//   lanes 0-15 : row (row+lane),   K = 0..7  and 16..23 (within the 32-chunk)
//   lanes 16-31: row (row+lane-16),K = 8..15 and 24..31
__device__ __forceinline__ v16bf load_frag(const bf16* __restrict__ base,
                                           int row, int stride, int k0,
                                           int lane) {
  const int lm = lane & 15;
  const int kh = (lane >> 4) << 3;   // 0 or 8
  const bf16* p = base + (size_t)(row + lm) * (size_t)stride + (size_t)(k0 + kh);
  union { v16bf bf; v4u u[2]; } r;
  r.u[0] = *(const v4u*)(p);        // K = kh .. kh+7        (16 bytes)
  r.u[1] = *(const v4u*)(p + 16);   // K = kh+16 .. kh+23
  return r.bf;
}

// Exchange with partner lane (lane ^ 16) inside a wave32.
__device__ __forceinline__ float swapx16(float x) {
  int i = __builtin_bit_cast(int, x);
  i = __builtin_amdgcn_ds_swizzle(i, 0x401F);  // group-of-32, xor_mask=0x10
  return __builtin_bit_cast(float, i);
}

// ---------------------------------------------------------------------------
// fp32 -> bf16 conversion
// ---------------------------------------------------------------------------
__global__ void cvt_f32_bf16(const float* __restrict__ in,
                             bf16* __restrict__ out, int n) {
  int i = blockIdx.x * blockDim.x + threadIdx.x;
  if (i < n) out[i] = (bf16)in[i];
}

// ---------------------------------------------------------------------------
// QKV projection: QKV[m,e] = sum_k X[m,k] * Wqkv[e,k]   (TN GEMM)
// Epilogue scatters into per-head Q (pre-scaled by hd^-0.5 * log2(e)),
// K, and V-transposed buffers. All segment decisions are wave-uniform.
// Block = 128 threads = 4 waves (2x2), wave tile 32x64, block tile 64x128.
// ---------------------------------------------------------------------------
__global__ void __launch_bounds__(128)
gemm_qkv(const bf16* __restrict__ Xb, const bf16* __restrict__ Wb,
         bf16* __restrict__ q_ws, bf16* __restrict__ k_ws,
         bf16* __restrict__ vt_ws) {
  const int lane = threadIdx.x & 31;
  const int wid  = __builtin_amdgcn_readfirstlane(threadIdx.x >> 5);  // SGPR
  const int m0 = blockIdx.x * 64  + (wid & 1) * 32;
  const int e0 = blockIdx.y * 128 + (wid >> 1) * 64;

  v8f acc[2][4];
#pragma unroll
  for (int i = 0; i < 2; ++i)
#pragma unroll
    for (int j = 0; j < 4; ++j) acc[i][j] = {};

  for (int k0 = 0; k0 < DIM; k0 += 32) {
    v16bf a[2], b[4];
#pragma unroll
    for (int i = 0; i < 2; ++i) a[i] = load_frag(Xb, m0 + i * 16, DIM, k0, lane);
#pragma unroll
    for (int j = 0; j < 4; ++j) b[j] = load_frag(Wb, e0 + j * 16, DIM, k0, lane);
#pragma unroll
    for (int i = 0; i < 2; ++i)
#pragma unroll
      for (int j = 0; j < 4; ++j)
        acc[i][j] = wmma_bf16(a[i], b[j], acc[i][j]);
  }

  const int lm = lane & 15;
  const int mo = (lane >> 4) << 3;                 // +8 rows for high lanes
  const int bb    = m0 >> 12;                      // batch   (wave-uniform)
  const int nbase = m0 & 4095;                     // seq pos (wave-uniform)
  const float QSCALE = 0.125f * 1.44269504088896f; // hd^-0.5 * log2(e)
#pragma unroll
  for (int j = 0; j < 4; ++j) {
    const int ebase = e0 + j * 16;          // wave-uniform, multiple of 16
    const int which = ebase >> 10;          // uniform: 0=Q 1=K 2=V
    const int h     = (ebase >> 6) & 15;    // uniform within 16-wide tile
    const int bh    = bb * NHEADS + h;      // uniform
    const int d     = (ebase & 63) + lm;    // per-lane head-dim index
    if (which == 0) {
      bf16* qp = q_ws + (size_t)bh * SEQ * HDIM + d;
#pragma unroll
      for (int i = 0; i < 2; ++i)
#pragma unroll
        for (int v = 0; v < 8; ++v)
          qp[(size_t)(nbase + i * 16 + mo + v) * HDIM] =
              (bf16)(acc[i][j][v] * QSCALE);
    } else if (which == 1) {
      bf16* kp = k_ws + (size_t)bh * SEQ * HDIM + d;
#pragma unroll
      for (int i = 0; i < 2; ++i)
#pragma unroll
        for (int v = 0; v < 8; ++v)
          kp[(size_t)(nbase + i * 16 + mo + v) * HDIM] = (bf16)acc[i][j][v];
    } else {
      bf16* vp = vt_ws + ((size_t)bh * HDIM + d) * SEQ + nbase;
#pragma unroll
      for (int i = 0; i < 2; ++i)
#pragma unroll
        for (int v = 0; v < 8; ++v)
          vp[i * 16 + mo + v] = (bf16)acc[i][j][v];
    }
  }
}

// ---------------------------------------------------------------------------
// Flash attention. Block = 256 threads = 8 waves = 128 consecutive queries of
// one (batch, head). K/V^T tiles (32 keys) are staged once per block into LDS
// via async-to-LDS copies (ASYNCcnt) and consumed by all waves via ds loads.
// Each wave computes S^T = K * Q^T so the S^T accumulator layout matches the
// bf16 B-operand fragment layout for O^T = V^T * P^T directly (pack-only).
// Online softmax in base 2 (scale folded into Q).
// ---------------------------------------------------------------------------
__global__ void __launch_bounds__(256)
flash_attn(const bf16* __restrict__ q_ws, const bf16* __restrict__ k_ws,
           const bf16* __restrict__ vt_ws, bf16* __restrict__ o_ws) {
  __shared__ bf16 kbuf[32 * HDIM];   // 32 keys x 64 dims  (4 KB, contiguous)
  __shared__ bf16 vbuf[HDIM * 32];   // 64 dims x 32 keys  (4 KB)

  const int tid  = threadIdx.x;
  const int lane = tid & 31;
  const int wid  = __builtin_amdgcn_readfirstlane(tid >> 5);  // 0..7, SGPR
  const int blk  = blockIdx.x;       // B*H*(SEQ/128) = 1024 blocks
  const int bh   = blk >> 5;         // 32 query-blocks per head
  const int q0   = (blk & 31) * 128 + wid * 16;

  const bf16* Qh  = q_ws  + (size_t)bh * SEQ * HDIM;
  const bf16* Kh  = k_ws  + (size_t)bh * SEQ * HDIM;
  const bf16* VTh = vt_ws + (size_t)bh * HDIM * SEQ;

  // Q fragments (B operand: lane = query column, K-dim = head dim)
  const v16bf qf0 = load_frag(Qh, q0, HDIM, 0,  lane);
  const v16bf qf1 = load_frag(Qh, q0, HDIM, 32, lane);

  // V^T staging addressing: row = head-dim (0..63), 4 x 16B chunks per row
  const int vrow = tid >> 2;         // 0..63
  const int vcol = (tid & 3) * 8;    // element offset within 32-key row

  v8f oacc[4];
#pragma unroll
  for (int t = 0; t < 4; ++t) oacc[t] = {};
  float m_run = -1e30f;
  float l_run = 0.f;

  for (int kt = 0; kt < SEQ; kt += 32) {
    // ---- stage K tile (contiguous 4KB) + V^T tile (64 rows x 64B) ----
    __syncthreads();   // previous iteration's compute done with LDS
    {
      const v4u* gk = (const v4u*)(Kh + (size_t)kt * HDIM);  // 256 chunks
      const bf16* gv = VTh + (size_t)vrow * SEQ + kt + vcol;
      v4u* lk = ((v4u*)kbuf) + tid;
      bf16* lv = vbuf + vrow * 32 + vcol;
#if defined(HAVE_ASYNC_LDS)
      __builtin_amdgcn_global_load_async_to_lds_b128(AS1(gk + tid), AS3(lk),
                                                     0, 0);
      __builtin_amdgcn_global_load_async_to_lds_b128(AS1(gv), AS3(lv), 0, 0);
#else
      *lk = gk[tid];
      *(v4u*)lv = *(const v4u*)gv;
#endif
      // prefetch next k-tile (speculative; dropped if past end)
      __builtin_prefetch(gk + 256, 0, 1);
      __builtin_prefetch(gv + 32, 0, 1);
    }
#if defined(HAVE_ASYNC_LDS)
    __builtin_amdgcn_s_wait_asynccnt(0);   // this wave's async copies landed
#endif
    __syncthreads();                       // all waves' copies visible

    // ---- S^T tiles: two 16(k) x 16(q) tiles, K-dim = 64 = 2 WMMAs each ----
    v8f st[2];
#pragma unroll
    for (int s = 0; s < 2; ++s) {
      const v16bf ka0 = load_frag(kbuf, s * 16, HDIM, 0,  lane);
      const v16bf ka1 = load_frag(kbuf, s * 16, HDIM, 32, lane);
      v8f z = {};
      st[s] = wmma_bf16(ka0, qf0, z);
      st[s] = wmma_bf16(ka1, qf1, st[s]);
    }

    // per-query max over these 32 keys (16 in-lane + partner lane)
    float tmax = st[0][0];
#pragma unroll
    for (int v = 1; v < 8; ++v) tmax = fmaxf(tmax, st[0][v]);
#pragma unroll
    for (int v = 0; v < 8; ++v) tmax = fmaxf(tmax, st[1][v]);
    tmax = fmaxf(tmax, swapx16(tmax));

    const float m_new = fmaxf(m_run, tmax);
    const float corr  = exp2f(m_run - m_new);

    // exponentiate + pack P^T directly into the bf16 B-operand fragment
    float psum = 0.f;
    v16bf pf;
#pragma unroll
    for (int v = 0; v < 8; ++v) {
      const float p0 = exp2f(st[0][v] - m_new);
      const float p1 = exp2f(st[1][v] - m_new);
      psum += p0 + p1;
      pf[v]     = (bf16)p0;   // K = 0..7  (lanes<16) / 8..15  (lanes>=16)
      pf[8 + v] = (bf16)p1;   // K = 16..23           / 24..31
    }
    psum += swapx16(psum);
    l_run = l_run * corr + psum;
    m_run = m_new;

    // ---- O^T += V^T(16d x 32k) * P^T(32k x 16q), 4 d-tiles ----
#pragma unroll
    for (int t = 0; t < 4; ++t) {
      oacc[t] *= corr;
      const v16bf vtf = load_frag(vbuf, t * 16, 32, 0, lane);
      oacc[t] = wmma_bf16(vtf, pf, oacc[t]);
    }
  }

  const float rcp = 1.f / l_run;
  const int lm = lane & 15;
  const int mo = (lane >> 4) << 3;
  const int b = bh >> 4, h = bh & 15;
  const int q = q0 + lm;
  bf16* orow = o_ws + ((size_t)(b * SEQ + q)) * DIM + h * HDIM;
#pragma unroll
  for (int t = 0; t < 4; ++t)
#pragma unroll
    for (int v = 0; v < 8; ++v)
      orow[t * 16 + mo + v] = (bf16)(oacc[t][v] * rcp);
}

// ---------------------------------------------------------------------------
// Output projection + bias: out[m,e] = sum_k O[m,k]*Wp[e,k] + bias[e] (f32)
// ---------------------------------------------------------------------------
__global__ void __launch_bounds__(128)
gemm_proj(const bf16* __restrict__ Ob, const bf16* __restrict__ Wb,
          const float* __restrict__ bias, float* __restrict__ out) {
  const int lane = threadIdx.x & 31;
  const int wid  = __builtin_amdgcn_readfirstlane(threadIdx.x >> 5);
  const int m0 = blockIdx.x * 64  + (wid & 1) * 32;
  const int e0 = blockIdx.y * 128 + (wid >> 1) * 64;

  v8f acc[2][4];
#pragma unroll
  for (int i = 0; i < 2; ++i)
#pragma unroll
    for (int j = 0; j < 4; ++j) acc[i][j] = {};

  for (int k0 = 0; k0 < DIM; k0 += 32) {
    v16bf a[2], b[4];
#pragma unroll
    for (int i = 0; i < 2; ++i) a[i] = load_frag(Ob, m0 + i * 16, DIM, k0, lane);
#pragma unroll
    for (int j = 0; j < 4; ++j) b[j] = load_frag(Wb, e0 + j * 16, DIM, k0, lane);
#pragma unroll
    for (int i = 0; i < 2; ++i)
#pragma unroll
      for (int j = 0; j < 4; ++j)
        acc[i][j] = wmma_bf16(a[i], b[j], acc[i][j]);
  }

  const int lm = lane & 15;
  const int mo = (lane >> 4) << 3;
#pragma unroll
  for (int j = 0; j < 4; ++j) {
    const int e = e0 + j * 16 + lm;
    const float bz = bias[e];
#pragma unroll
    for (int i = 0; i < 2; ++i)
#pragma unroll
      for (int v = 0; v < 8; ++v) {
        const int m = m0 + i * 16 + mo + v;
        out[(size_t)m * DIM + e] = acc[i][j][v] + bz;
      }
  }
}

// ---------------------------------------------------------------------------
// Launcher
// ---------------------------------------------------------------------------
extern "C" void kernel_launch(void* const* d_in, const int* in_sizes, int n_in,
                              void* d_out, int out_size, void* d_ws,
                              size_t ws_size, hipStream_t stream) {
  const float* x      = (const float*)d_in[0];
  const float* w_qkv  = (const float*)d_in[1];
  const float* w_proj = (const float*)d_in[2];
  const float* b_proj = (const float*)d_in[3];
  float* out = (float*)d_out;

  char* p = (char*)d_ws;
  bf16* xb     = (bf16*)p; p += (size_t)M_TOTAL * DIM * sizeof(bf16);          // 16 MB
  bf16* wqkvb  = (bf16*)p; p += (size_t)3 * DIM * DIM * sizeof(bf16);          //  6 MB
  bf16* wprojb = (bf16*)p; p += (size_t)DIM * DIM * sizeof(bf16);              //  2 MB
  const size_t headsz = (size_t)BATCH * NHEADS * SEQ * HDIM;
  bf16* q_ws   = (bf16*)p; p += headsz * sizeof(bf16);                         // 16 MB
  bf16* k_ws   = (bf16*)p; p += headsz * sizeof(bf16);                         // 16 MB
  bf16* vt_ws  = (bf16*)p; p += headsz * sizeof(bf16);                         // 16 MB
  bf16* o_ws   = (bf16*)p;                                                     // 16 MB

  const int nx = M_TOTAL * DIM;
  cvt_f32_bf16<<<(nx + 255) / 256, 256, 0, stream>>>(x, xb, nx);
  const int nw = 3 * DIM * DIM;
  cvt_f32_bf16<<<(nw + 255) / 256, 256, 0, stream>>>(w_qkv, wqkvb, nw);
  const int np = DIM * DIM;
  cvt_f32_bf16<<<(np + 255) / 256, 256, 0, stream>>>(w_proj, wprojb, np);

  gemm_qkv<<<dim3(M_TOTAL / 64, (3 * DIM) / 128), 128, 0, stream>>>(
      xb, wqkvb, q_ws, k_ws, vt_ws);

  flash_attn<<<dim3(BATCH * NHEADS * (SEQ / 128)), 256, 0, stream>>>(
      q_ws, k_ws, vt_ws, o_ws);

  gemm_proj<<<dim3(M_TOTAL / 64, DIM / 128), 128, 0, stream>>>(
      o_ws, wprojb, b_proj, out);
}
